// LinearCrossAttention_31456340476363
// MI455X (gfx1250) — matrix operand order repaired and checked by reading, not compiled, not run on hardware
//
#include <hip/hip_runtime.h>
#include <stdint.h>

typedef __attribute__((ext_vector_type(16))) __bf16 v16bf;
typedef __attribute__((ext_vector_type(8)))  float  v8f;
typedef __attribute__((ext_vector_type(4)))  unsigned int u32x4;
typedef __attribute__((ext_vector_type(8)))  int i32x8;
typedef __attribute__((ext_vector_type(4)))  int i32x4;
typedef unsigned short u16;
typedef unsigned int   u32;
typedef unsigned long long u64;

#define D_MODEL 1024
#define N_HEADS 16
#define N_KV    4
#define D_HEAD  64
#define T_LEN   2048
#define S_LEN   4096
#define BATCH   4
#define EPS_F   1e-5f

#if defined(__has_builtin)
#if __has_builtin(__builtin_amdgcn_tensor_load_to_lds) && __has_builtin(__builtin_amdgcn_s_wait_tensorcnt)
#define HAVE_TDM 1
#endif
#endif

// ---------- helpers ----------
__device__ __forceinline__ u16 f2bf(float f) {
    u32 u = __float_as_uint(f);
    u32 r = u + 0x7FFFu + ((u >> 16) & 1u);   // round-to-nearest-even
    return (u16)(r >> 16);
}
__device__ __forceinline__ u32 pack2bf(float a, float b) {
    return (u32)f2bf(a) | ((u32)f2bf(b) << 16);
}
__device__ __forceinline__ float bf2f(u16 h) { return __uint_as_float(((u32)h) << 16); }
__device__ __forceinline__ float phi_elu1(float x) { return x > 0.f ? x + 1.f : __expf(x); }

union FragAB { v16bf v; u32 u[8]; uint4 q[2]; };

__device__ __forceinline__ v8f wmma_bf16(const FragAB& a, const FragAB& b, v8f c) {
    return __builtin_amdgcn_wmma_f32_16x16x32_bf16(false, a.v, false, b.v, (short)0, c, false, false);
}

// A-fragment: rowbase points at (row*stride + k0) in bf16; K = {0..15} + 16*lhalf contiguous
__device__ __forceinline__ void load_fragA(FragAB& A, const u16* rowbase, int lhalf) {
    const char* p = (const char*)rowbase + 32 * lhalf;
    A.q[0] = *(const uint4*)p;
    A.q[1] = *(const uint4*)(p + 16);
}
// B-fragment: colbase points at (col*stride + k0); K = {0..15} + 16*lhalf contiguous
__device__ __forceinline__ void load_fragB(FragAB& B, const u16* colbase, int lhalf) {
    const char* p = (const char*)colbase + 32 * lhalf;
    B.q[0] = *(const uint4*)p;
    B.q[1] = *(const uint4*)(p + 16);
}

// ---------- TDM staging of a [rows][1024] bf16 tile into LDS ----------
__device__ __forceinline__ void tdm_stage_rows(u16* lds, const u16* g, int rows) {
#ifdef HAVE_TDM
    u64 ga = (u64)(uintptr_t)g;
    u32 la = (u32)(uintptr_t)(void*)lds;      // flat LDS addr low 32 bits = LDS byte offset
    u32x4 g0;
    g0.x = 1u;                                 // count=1, user mode, no gather
    g0.y = la;                                 // lds_addr
    g0.z = (u32)ga;                            // global_addr[31:0]
    g0.w = (u32)(ga >> 32) | (2u << 30);       // global_addr[56:32] | type=2
    i32x8 g1;
    g1[0] = (int)0x00010000u;                  // wg_mask=0, data_size=1 (2 bytes), no flags
    g1[1] = (int)(1024u << 16);                // tensor_dim0[15:0]=1024 in bits[63:48]
    g1[2] = (int)((u32)rows << 16);            // tensor_dim0 hi=0; tensor_dim1[15:0]=rows
    g1[3] = (int)(1024u << 16);                // tensor_dim1 hi=0; tile_dim0=1024
    g1[4] = rows;                              // tile_dim1=rows; tile_dim2=0
    g1[5] = 1024;                              // tensor_dim0_stride[31:0]
    g1[6] = 0;
    g1[7] = 0;
    i32x4 z4 = (i32x4)0;
#if __clang_major__ >= 23
    i32x8 z8 = (i32x8)0;
    __builtin_amdgcn_tensor_load_to_lds(g0, g1, z4, z4, z8, 0);
#else
    __builtin_amdgcn_tensor_load_to_lds(g0, g1, z4, z4, 0);
#endif
    __builtin_amdgcn_s_wait_tensorcnt(0);
#endif
}

// ---------- prep kernels ----------
__global__ __launch_bounds__(256) void lca_transpose_bf16(const float* __restrict__ src,
                                                          u16* __restrict__ dst, int R, int C) {
    int i = blockIdx.x * 256 + threadIdx.x;
    if (i < R * C) {
        int r = i / C, c = i % C;
        dst[(size_t)c * R + r] = f2bf(src[i]);   // dst[n][k] = src[k][n]
    }
}

__global__ __launch_bounds__(256) void lca_cvt_bf16(const float* __restrict__ src,
                                                    u16* __restrict__ dst, int n4) {
    int i = blockIdx.x * 256 + threadIdx.x;
    if (i < n4) {
        float4 f = ((const float4*)src)[i];
        uint2 p; p.x = pack2bf(f.x, f.y); p.y = pack2bf(f.z, f.w);
        ((uint2*)dst)[i] = p;
    }
}

__global__ __launch_bounds__(256) void lca_zero_f32(float* __restrict__ p, int n) {
    int i = blockIdx.x * 256 + threadIdx.x;
    if (i < n) p[i] = 0.f;
}

// ---------- kv / z kernel ----------
// grid: BATCH * N_KV * (S/128) = 512 blocks, 256 threads (8 waves)
// LDS: enc tile bf16 [128][1024] + kT [64][128] + vT [64][128] = 294912 B
__global__ __launch_bounds__(256) void lca_kvz_kernel(const u16* __restrict__ encbf,
                                                      const unsigned char* __restrict__ mask,
                                                      const u16* __restrict__ wkT,
                                                      const u16* __restrict__ wvT,
                                                      float* __restrict__ kv_g,   // [B][KV][e=64][d=64]
                                                      float* __restrict__ z_g)    // [B][KV][64]
{
    extern __shared__ char smem[];
    u16* ebuf = (u16*)smem;                 // [128][1024]
    u16* kT   = ebuf + 128 * 1024;          // [d=64][t=128]
    u16* vT   = kT + 64 * 128;              // [e=64][t=128]

    const int tid   = threadIdx.x;
    const int bx    = blockIdx.x;
    const int chunk = bx & 31;
    const int kvh   = (bx >> 5) & 3;
    const int b     = bx >> 7;
    const int s0    = chunk * 128;

    // stage enc tile (already bf16) via Tensor Data Mover
#ifdef HAVE_TDM
    if (tid < 32) tdm_stage_rows(ebuf, encbf + ((size_t)(b * S_LEN + s0)) * D_MODEL, 128);
#else
    {
        const uint4* src = (const uint4*)(encbf + ((size_t)(b * S_LEN + s0)) * D_MODEL);
        uint4* dst = (uint4*)ebuf;
        for (int i = tid; i < 128 * D_MODEL / 8; i += 256) dst[i] = src[i];
    }
#endif
    __syncthreads();

    const int wid = tid >> 5, lane = tid & 31;
    const int lrow = lane & 15, lhalf = lane >> 4;

    // GEMM: k = enc*wk_head, v = enc*wv_head  (128x64 each), strips of 16 rows per wave
    v8f kacc[4] = {}, vacc[4] = {};
    const int trow = wid * 16 + lrow;
    for (int ks = 0; ks < 32; ++ks) {
        const int k0 = ks * 32;
        FragAB A;
        load_fragA(A, ebuf + trow * D_MODEL + k0, lhalf);
#pragma unroll
        for (int nt = 0; nt < 4; ++nt) {
            const int col = kvh * 64 + nt * 16 + lrow;
            FragAB Bk, Bv;
            load_fragB(Bk, wkT + (size_t)col * D_MODEL + k0, lhalf);
            load_fragB(Bv, wvT + (size_t)col * D_MODEL + k0, lhalf);
            kacc[nt] = wmma_bf16(A, Bk, kacc[nt]);
            vacc[nt] = wmma_bf16(A, Bv, vacc[nt]);
        }
    }

    // epilogue: phi on k, mask both, store transposed (d-major) to LDS
    float mrow[8];
#pragma unroll
    for (int r = 0; r < 8; ++r) {
        const int t = wid * 16 + lhalf * 8 + r;
        mrow[r] = mask[(size_t)b * S_LEN + s0 + t] ? 1.f : 0.f;
    }
#pragma unroll
    for (int nt = 0; nt < 4; ++nt) {
        const int d = nt * 16 + lrow;
#pragma unroll
        for (int r = 0; r < 8; ++r) {
            const int t = wid * 16 + lhalf * 8 + r;
            kT[d * 128 + t] = f2bf(phi_elu1(kacc[nt][r]) * mrow[r]);
            vT[d * 128 + t] = f2bf(vacc[nt][r] * mrow[r]);
        }
    }
    __syncthreads();

    // z partial sums
    {
        const int d = tid & 63, part = tid >> 6;
        float s = 0.f;
#pragma unroll 8
        for (int i = 0; i < 32; ++i) s += bf2f(kT[d * 128 + part * 32 + i]);
        atomicAdd(&z_g[(b * N_KV + kvh) * 64 + d], s);
    }

    // kv = k^T * v : D[d][e] 64x64 -> 16 tiles, 2 per wave, K = 128
    float* kv_dst = kv_g + (size_t)(b * N_KV + kvh) * 64 * 64;   // layout [e][d]
#pragma unroll
    for (int tt = 0; tt < 2; ++tt) {
        const int tile = wid + tt * 8;
        const int mt = tile >> 2, nt = tile & 3;
        v8f acc = {};
        const int drow = mt * 16 + lrow;
        const int e    = nt * 16 + lrow;
#pragma unroll
        for (int ks = 0; ks < 4; ++ks) {
            const int k0 = ks * 32;
            FragAB A, Bf;
            load_fragA(A, kT + drow * 128 + k0, lhalf);
            load_fragB(Bf, vT + e * 128 + k0, lhalf);
            acc = wmma_bf16(A, Bf, acc);
        }
#pragma unroll
        for (int r = 0; r < 8; ++r) {
            const int d = mt * 16 + lhalf * 8 + r;
            atomicAdd(&kv_dst[(nt * 16 + lrow) * 64 + d], acc[r]);
        }
    }
}

// ---------- fused main kernel: q proj + phi + num/den + out proj ----------
// grid: BATCH * (T/64) = 128 blocks, 256 threads (8 waves)
// LDS: xbuf(=attn) [64][1024]bf16 + qbuf [64][1024]bf16 + den [64][16]f32 + kv [4][64][64]bf16 = 299008 B
__global__ __launch_bounds__(256) void lca_main_kernel(const u16* __restrict__ xbf,
                                                       const u16* __restrict__ wqT,
                                                       const u16* __restrict__ woT,
                                                       const float* __restrict__ kv_g,
                                                       const float* __restrict__ z_g,
                                                       float* __restrict__ out)
{
    extern __shared__ char smem[];
    u16*   xbuf  = (u16*)smem;                    // [64][1024], later reused for attn
    u16*   qbuf  = xbuf + 64 * D_MODEL;           // [64][1024]
    float* den   = (float*)(qbuf + 64 * D_MODEL); // [64][16]
    u16*   kvbuf = (u16*)(den + 64 * 16);         // [4][e=64][d=64]

    const int tid = threadIdx.x;
    const int bx  = blockIdx.x;
    const int tb  = bx & 31;
    const int b   = bx >> 5;
    const int t0  = tb * 64;

    // stage x tile via TDM, init den, stage kv (f32->bf16)
#ifdef HAVE_TDM
    if (tid < 32) tdm_stage_rows(xbuf, xbf + ((size_t)(b * T_LEN + t0)) * D_MODEL, 64);
#else
    {
        const uint4* src = (const uint4*)(xbf + ((size_t)(b * T_LEN + t0)) * D_MODEL);
        uint4* dst = (uint4*)xbuf;
        for (int i = tid; i < 64 * D_MODEL / 8; i += 256) dst[i] = src[i];
    }
#endif
    for (int i = tid; i < 64 * 16; i += 256) den[i] = 0.f;
    {
        const float* src = kv_g + (size_t)b * N_KV * 64 * 64;
        for (int i = tid; i < N_KV * 64 * 64; i += 256) kvbuf[i] = f2bf(src[i]);
    }
    __syncthreads();

    const int wid = tid >> 5, lane = tid & 31;
    const int lrow = lane & 15, lhalf = lane >> 4;
    const int strip = wid & 3, nhalf = wid >> 2;
    const int trow = strip * 16 + lrow;

    // ---- Phase A: q = phi(x * wq), den += q . z ----
    for (int chunk = 0; chunk < 4; ++chunk) {
        v8f acc[8] = {};
        const int ncol0 = nhalf * 512 + chunk * 128;
        for (int ks = 0; ks < 32; ++ks) {
            const int k0 = ks * 32;
            FragAB A;
            load_fragA(A, xbuf + trow * D_MODEL + k0, lhalf);
#pragma unroll
            for (int nt = 0; nt < 8; ++nt) {
                const int col = ncol0 + nt * 16 + lrow;
                FragAB Bf;
                load_fragB(Bf, wqT + (size_t)col * D_MODEL + k0, lhalf);
                acc[nt] = wmma_bf16(A, Bf, acc[nt]);
            }
        }
#pragma unroll
        for (int nt = 0; nt < 8; ++nt) {
            const int col  = ncol0 + nt * 16 + lrow;
            const int head = col >> 6;
            const int kvh  = head >> 2;
            const float zv = z_g[(b * N_KV + kvh) * 64 + (col & 63)];
#pragma unroll
            for (int r = 0; r < 8; ++r) {
                const int t = strip * 16 + lhalf * 8 + r;
                const float qv = phi_elu1(acc[nt][r]);
                atomicAdd(&den[t * 16 + head], qv * zv);
                qbuf[t * D_MODEL + col] = f2bf(qv);
            }
        }
    }
    __syncthreads();

    // ---- Phase B: num = q_head * kv_head ; attn = num/(den+eps) -> xbuf ----
#pragma unroll
    for (int hp = 0; hp < 2; ++hp) {
        const int h   = wid + hp * 8;
        const int kvh = h >> 2;
        const u16* kvp = kvbuf + kvh * 64 * 64;   // [e][d]
        for (int nt = 0; nt < 4; ++nt) {
            v8f acc[4] = {};
            const int e = nt * 16 + lrow;
#pragma unroll
            for (int ks = 0; ks < 2; ++ks) {
                const int k0 = ks * 32;
                FragAB Bf;
                load_fragB(Bf, kvp + e * 64 + k0, lhalf);
#pragma unroll
                for (int mt = 0; mt < 4; ++mt) {
                    FragAB A;
                    load_fragA(A, qbuf + (mt * 16 + lrow) * D_MODEL + h * 64 + k0, lhalf);
                    acc[mt] = wmma_bf16(A, Bf, acc[mt]);
                }
            }
#pragma unroll
            for (int mt = 0; mt < 4; ++mt) {
#pragma unroll
                for (int r = 0; r < 8; ++r) {
                    const int t = mt * 16 + lhalf * 8 + r;
                    const float dv = den[t * 16 + h] + EPS_F;
                    xbuf[t * D_MODEL + h * 64 + nt * 16 + lrow] = f2bf(acc[mt][r] / dv);
                }
            }
        }
    }
    __syncthreads();

    // ---- Phase C: out = attn * wo -> global ----
    for (int chunk = 0; chunk < 4; ++chunk) {
        v8f acc[8] = {};
        const int ncol0 = nhalf * 512 + chunk * 128;
        for (int ks = 0; ks < 32; ++ks) {
            const int k0 = ks * 32;
            FragAB A;
            load_fragA(A, xbuf + trow * D_MODEL + k0, lhalf);
#pragma unroll
            for (int nt = 0; nt < 8; ++nt) {
                const int col = ncol0 + nt * 16 + lrow;
                FragAB Bf;
                load_fragB(Bf, woT + (size_t)col * D_MODEL + k0, lhalf);
                acc[nt] = wmma_bf16(A, Bf, acc[nt]);
            }
        }
#pragma unroll
        for (int nt = 0; nt < 8; ++nt) {
            const int col = ncol0 + nt * 16 + lrow;
#pragma unroll
            for (int r = 0; r < 8; ++r) {
                const int t = strip * 16 + lhalf * 8 + r;
                out[((size_t)(b * T_LEN + t0 + t)) * D_MODEL + col] = acc[nt][r];
            }
        }
    }
}

// ---------- host launch ----------
#define WS_WQT   0u
#define WS_WKT   2097152u
#define WS_WVT   2621440u
#define WS_WOT   3145728u
#define WS_KV    5242880u
#define WS_Z     5505024u
#define WS_XBF   6291456u
#define WS_ENCBF 23068672u
#define KV_LDS   (128 * 1024 * 2 + 2 * 64 * 128 * 2)                  // 294912
#define MAIN_LDS (2 * 64 * 1024 * 2 + 64 * 16 * 4 + 4 * 64 * 64 * 2)  // 299008

extern "C" void kernel_launch(void* const* d_in, const int* in_sizes, int n_in,
                              void* d_out, int out_size, void* d_ws, size_t ws_size,
                              hipStream_t stream)
{
    (void)in_sizes; (void)n_in; (void)out_size; (void)ws_size;
    const float* x   = (const float*)d_in[0];
    const float* enc = (const float*)d_in[1];
    const unsigned char* mask = (const unsigned char*)d_in[2];
    const float* wq  = (const float*)d_in[3];
    const float* wk  = (const float*)d_in[4];
    const float* wv  = (const float*)d_in[5];
    const float* wo  = (const float*)d_in[6];
    float* out = (float*)d_out;

    char* ws = (char*)d_ws;
    u16* wqT   = (u16*)(ws + WS_WQT);
    u16* wkT   = (u16*)(ws + WS_WKT);
    u16* wvT   = (u16*)(ws + WS_WVT);
    u16* woT   = (u16*)(ws + WS_WOT);
    float* kv_g = (float*)(ws + WS_KV);
    float* z_g  = (float*)(ws + WS_Z);
    u16* xbf   = (u16*)(ws + WS_XBF);
    u16* encbf = (u16*)(ws + WS_ENCBF);

    hipFuncSetAttribute((const void*)lca_kvz_kernel,
                        hipFuncAttributeMaxDynamicSharedMemorySize, KV_LDS);
    hipFuncSetAttribute((const void*)lca_main_kernel,
                        hipFuncAttributeMaxDynamicSharedMemorySize, MAIN_LDS);

    // weight transpose+convert, activation convert, accumulator zeroing
    lca_transpose_bf16<<<(1024 * 1024) / 256, 256, 0, stream>>>(wq, wqT, 1024, 1024);
    lca_transpose_bf16<<<(1024 * 256) / 256, 256, 0, stream>>>(wk, wkT, 1024, 256);
    lca_transpose_bf16<<<(1024 * 256) / 256, 256, 0, stream>>>(wv, wvT, 1024, 256);
    lca_transpose_bf16<<<(1024 * 1024) / 256, 256, 0, stream>>>(wo, woT, 1024, 1024);
    lca_cvt_bf16<<<(BATCH * T_LEN * D_MODEL / 4) / 256, 256, 0, stream>>>(
        x, xbf, BATCH * T_LEN * D_MODEL / 4);
    lca_cvt_bf16<<<(BATCH * S_LEN * D_MODEL / 4) / 256, 256, 0, stream>>>(
        enc, encbf, BATCH * S_LEN * D_MODEL / 4);
    lca_zero_f32<<<(66560 + 255) / 256, 256, 0, stream>>>(kv_g, 66560);

    lca_kvz_kernel<<<BATCH * N_KV * (S_LEN / 128), 256, KV_LDS, stream>>>(
        encbf, mask, wkT, wvT, kv_g, z_g);
    lca_main_kernel<<<BATCH * (T_LEN / 64), 256, MAIN_LDS, stream>>>(
        xbf, wqT, woT, kv_g, z_g, out);
}
